// LocalAttention_26156350833320
// MI455X (gfx1250) — compile-verified, matrix operand order checked by reading, not run on hardware
//
#include <hip/hip_runtime.h>

// Problem constants (from reference)
#define BATCH   2
#define LSEQ    2048
#define DMODEL  1024
#define NHEAD   16
#define DH      64
#define WIN     256

typedef __bf16 bf16_t;
typedef bf16_t bf16x4  __attribute__((ext_vector_type(4)));
typedef bf16_t bf16x8  __attribute__((ext_vector_type(8)));
typedef bf16_t bf16x16 __attribute__((ext_vector_type(16)));
typedef float  f32x4   __attribute__((ext_vector_type(4)));
typedef float  f32x8   __attribute__((ext_vector_type(8)));

// ---------------------------------------------------------------------------
// Kernel 0a: stream-convert hidden fp32 -> bf16 (packed cvt, 16B loads)
// ---------------------------------------------------------------------------
__global__ __launch_bounds__(256)
void cvt_x_kernel(const float* __restrict__ X, bf16_t* __restrict__ Xb, int n4)
{
    const int stride = gridDim.x * blockDim.x;
    for (int i = blockIdx.x * blockDim.x + threadIdx.x; i < n4; i += stride) {
        const f32x4 v = *(const f32x4*)(X + (size_t)i * 4);
        bf16x4 o;
        o[0] = (bf16_t)v[0]; o[1] = (bf16_t)v[1];
        o[2] = (bf16_t)v[2]; o[3] = (bf16_t)v[3];
        *(bf16x4*)(Xb + (size_t)i * 4) = o;
    }
}

// ---------------------------------------------------------------------------
// Kernel 0b: transpose + convert W (fp32 [k][c] -> bf16 Wt[c][k]) for Q/K/V.
// 32x32 LDS tiles, fully coalesced both sides. grid = (32, 32, 3).
// ---------------------------------------------------------------------------
__global__ __launch_bounds__(256)
void wt_kernel(const float* __restrict__ Wq, const float* __restrict__ Wk,
               const float* __restrict__ Wv, bf16_t* __restrict__ Wt)
{
    __shared__ float tile[32][33];
    const int which = blockIdx.z;
    const float* W = (which == 0) ? Wq : (which == 1) ? Wk : Wv;
    bf16_t* Out = Wt + (size_t)which * DMODEL * DMODEL;

    const int tx = threadIdx.x & 31;
    const int ty = threadIdx.x >> 5;          // 0..7
    const int c0 = blockIdx.x * 32;
    const int k0 = blockIdx.y * 32;

#pragma unroll
    for (int i = 0; i < 32; i += 8)
        tile[ty + i][tx] = W[(size_t)(k0 + ty + i) * DMODEL + c0 + tx];
    __syncthreads();
#pragma unroll
    for (int i = 0; i < 32; i += 8)
        Out[(size_t)(c0 + ty + i) * DMODEL + k0 + tx] = (bf16_t)tile[tx][ty + i];
}

// ---------------------------------------------------------------------------
// Kernel 1: QKV projection GEMM, all-bf16 operands, fp32 accumulate.
// grid = (16 col-tiles, 64 row-tiles, 3 {Q,K,V}), block = 128 (4 waves).
// Every fragment = two contiguous 16B loads per lane (A-layout runs).
// Q,K stored [B][H][L][dh]; V stored transposed [B][H][dh][L].
// ---------------------------------------------------------------------------
__global__ __launch_bounds__(128)
void qkv_proj_kernel(const bf16_t* __restrict__ Xb,
                     const bf16_t* __restrict__ Wt,   // 3 x [1024 cols][1024 k]
                     const float* __restrict__ bq, const float* __restrict__ bk,
                     const float* __restrict__ bv,
                     bf16_t* __restrict__ Qs, bf16_t* __restrict__ Ks,
                     bf16_t* __restrict__ Vt)
{
    const int lane = threadIdx.x & 31;
    const int wave = threadIdx.x >> 5;
    const int lcol = lane & 15;
    const int half = lane >> 4;
    const int kb8  = half * 8;

    const int which = blockIdx.z;
    const bf16_t* Wm   = Wt + (size_t)which * DMODEL * DMODEL;
    const float*  bias = (which == 0) ? bq : (which == 1) ? bk : bv;

    const int col0 = blockIdx.x * 64;
    const int row0 = blockIdx.y * 64 + wave * 16;

    f32x8 acc[4];
#pragma unroll
    for (int ns = 0; ns < 4; ++ns) {
        const float bv_ = bias[col0 + ns * 16 + lcol];
#pragma unroll
        for (int r = 0; r < 8; ++r) acc[ns][r] = bv_;
    }

    const bf16_t* xrow = Xb + (size_t)(row0 + lcol) * DMODEL;

    for (int kc = 0; kc < DMODEL; kc += 32) {
        bf16x16 a;
        {
            const bf16x8 lo = *(const bf16x8*)(xrow + kc + kb8);
            const bf16x8 hi = *(const bf16x8*)(xrow + kc + 16 + kb8);
#pragma unroll
            for (int e = 0; e < 8; ++e) { a[e] = lo[e]; a[8 + e] = hi[e]; }
        }
#pragma unroll
        for (int ns = 0; ns < 4; ++ns) {
            // B[k][n] = W[k][col] = Wt[col][k]  (k contiguous per lane)
            const bf16_t* wrow = Wm + (size_t)(col0 + ns * 16 + lcol) * DMODEL;
            bf16x16 bfr;
            const bf16x8 lo = *(const bf16x8*)(wrow + kc + kb8);
            const bf16x8 hi = *(const bf16x8*)(wrow + kc + 16 + kb8);
#pragma unroll
            for (int e = 0; e < 8; ++e) { bfr[e] = lo[e]; bfr[8 + e] = hi[e]; }
            acc[ns] = __builtin_amdgcn_wmma_f32_16x16x32_bf16(
                false, a, false, bfr, (short)0, acc[ns], false, false);
        }
    }

#pragma unroll
    for (int ns = 0; ns < 4; ++ns) {
        const int col = col0 + ns * 16 + lcol;
        const int h = col >> 6, d = col & (DH - 1);
#pragma unroll
        for (int r = 0; r < 8; ++r) {
            const int m = row0 + r + 8 * half;
            const int b = m >> 11, l = m & (LSEQ - 1);
            const bf16_t v = (bf16_t)acc[ns][r];
            if (which == 2) // V: transposed [B][H][dh][L] for fast P@V frags
                Vt[(((size_t)b * NHEAD + h) * DH + d) * LSEQ + l] = v;
            else if (which == 0)
                Qs[(((size_t)b * NHEAD + h) * LSEQ + l) * DH + d] = v;
            else
                Ks[(((size_t)b * NHEAD + h) * LSEQ + l) * DH + d] = v;
        }
    }
}

// ---------------------------------------------------------------------------
// Kernel 2: sliding-window flash attention.
// grid = (L/64, H, B), block = 128; each wave owns a 16-query tile.
// Per 32-key step: 4 WMMA (Q*K^T) + online softmax + LDS P-transpose +
// 4 WMMA (P*V). All fragments are contiguous b128 loads.
// ---------------------------------------------------------------------------
__global__ __launch_bounds__(128)
void local_attn_kernel(const bf16_t* __restrict__ Qs,
                       const bf16_t* __restrict__ Ks,
                       const bf16_t* __restrict__ Vt,
                       float* __restrict__ out)
{
    __shared__ bf16_t plds[4][16][32];

    const int lane = threadIdx.x & 31;
    const int wave = threadIdx.x >> 5;
    const int lcol = lane & 15;
    const int half = lane >> 4;
    const int kb8  = half * 8;

    const int h  = blockIdx.y;
    const int b  = blockIdx.z;
    const int q0 = blockIdx.x * 64 + wave * 16;

    const bf16_t* Qp  = Qs + ((size_t)b * NHEAD + h) * LSEQ * DH;
    const bf16_t* Kp  = Ks + ((size_t)b * NHEAD + h) * LSEQ * DH;
    const bf16_t* Vtp = Vt + ((size_t)b * NHEAD + h) * DH * LSEQ;

    bf16x16 qf[2];
    {
        const bf16_t* qrow = Qp + (size_t)(q0 + lcol) * DH;
#pragma unroll
        for (int kk = 0; kk < 2; ++kk) {
            const bf16x8 lo = *(const bf16x8*)(qrow + kk * 32 + kb8);
            const bf16x8 hi = *(const bf16x8*)(qrow + kk * 32 + 16 + kb8);
#pragma unroll
            for (int e = 0; e < 8; ++e) { qf[kk][e] = lo[e]; qf[kk][8 + e] = hi[e]; }
        }
    }

    f32x8 o[4];
#pragma unroll
    for (int ns = 0; ns < 4; ++ns)
#pragma unroll
        for (int r = 0; r < 8; ++r) o[ns][r] = 0.0f;

    float m_r[8], l_r[8];
#pragma unroll
    for (int r = 0; r < 8; ++r) { m_r[r] = -1e30f; l_r[r] = 0.0f; }

    int kbeg = q0 - (WIN - 1); if (kbeg < 0) kbeg = 0;
    const int kb0 = kbeg >> 5;
    const int kb1 = (q0 + 15) >> 5;
    const float scale = 0.125f;   // 1/sqrt(64)

    for (int kb = kb0; kb <= kb1; ++kb) {
        const int ks = kb << 5;

        __builtin_prefetch(Kp + (size_t)(ks + 32) * DH, 0, 0);
        __builtin_prefetch(Vtp + (size_t)lcol * LSEQ + ks + 32, 0, 0);

        // ---- scores: two 16x16 tiles over 32 keys ----
        f32x8 s[2];
#pragma unroll
        for (int t = 0; t < 2; ++t) {
#pragma unroll
            for (int r = 0; r < 8; ++r) s[t][r] = 0.0f;
            const bf16_t* krow = Kp + (size_t)(ks + t * 16 + lcol) * DH;
#pragma unroll
            for (int kk = 0; kk < 2; ++kk) {
                bf16x16 kf;
                const bf16x8 lo = *(const bf16x8*)(krow + kk * 32 + kb8);
                const bf16x8 hi = *(const bf16x8*)(krow + kk * 32 + 16 + kb8);
#pragma unroll
                for (int e = 0; e < 8; ++e) { kf[e] = lo[e]; kf[8 + e] = hi[e]; }
                s[t] = __builtin_amdgcn_wmma_f32_16x16x32_bf16(
                    false, qf[kk], false, kf, (short)0, s[t], false, false);
            }
        }

        // ---- scale + band mask (j <= i && j > i-WIN) ----
#pragma unroll
        for (int t = 0; t < 2; ++t) {
            const int j = ks + t * 16 + lcol;
#pragma unroll
            for (int r = 0; r < 8; ++r) {
                const int i = q0 + r + 8 * half;
                float v = s[t][r] * scale;
                if (j > i || j <= i - WIN) v = -1e30f;
                s[t][r] = v;
            }
        }

        // ---- online softmax (row reductions across 16-lane halves) ----
        float p0[8], p1[8], alpha[8];
#pragma unroll
        for (int r = 0; r < 8; ++r) {
            float t = fmaxf(s[0][r], s[1][r]);
            t = fmaxf(t, __shfl_xor(t, 1, 32));
            t = fmaxf(t, __shfl_xor(t, 2, 32));
            t = fmaxf(t, __shfl_xor(t, 4, 32));
            t = fmaxf(t, __shfl_xor(t, 8, 32));
            const float mnew = fmaxf(m_r[r], t);
            alpha[r] = __expf(m_r[r] - mnew);
            p0[r] = __expf(s[0][r] - mnew);
            p1[r] = __expf(s[1][r] - mnew);
            float rs = p0[r] + p1[r];
            rs += __shfl_xor(rs, 1, 32);
            rs += __shfl_xor(rs, 2, 32);
            rs += __shfl_xor(rs, 4, 32);
            rs += __shfl_xor(rs, 8, 32);
            l_r[r] = l_r[r] * alpha[r] + rs;
            m_r[r] = mnew;
        }
#pragma unroll
        for (int ns = 0; ns < 4; ++ns)
#pragma unroll
            for (int r = 0; r < 8; ++r) o[ns][r] *= alpha[r];

        // ---- transpose P via per-wave LDS: C-layout -> 16-bit A-layout ----
#pragma unroll
        for (int r = 0; r < 8; ++r) {
            plds[wave][r + 8 * half][lcol]      = (bf16_t)p0[r];
            plds[wave][r + 8 * half][16 + lcol] = (bf16_t)p1[r];
        }
        asm volatile("s_wait_dscnt 0" ::: "memory");

        bf16x16 pa;
        {
            const bf16x8 lo = *(const bf16x8*)&plds[wave][lcol][kb8];
            const bf16x8 hi = *(const bf16x8*)&plds[wave][lcol][16 + kb8];
#pragma unroll
            for (int e = 0; e < 8; ++e) { pa[e] = lo[e]; pa[8 + e] = hi[e]; }
        }

        // ---- O += P @ V : B[k][n] = Vt[nbase+n][ks+k], k contiguous ----
#pragma unroll
        for (int ns = 0; ns < 4; ++ns) {
            const bf16_t* vrow = Vtp + (size_t)(ns * 16 + lcol) * LSEQ + ks;
            bf16x16 vf;
            const bf16x8 lo = *(const bf16x8*)(vrow + kb8);
            const bf16x8 hi = *(const bf16x8*)(vrow + 16 + kb8);
#pragma unroll
            for (int e = 0; e < 8; ++e) { vf[e] = lo[e]; vf[8 + e] = hi[e]; }
            o[ns] = __builtin_amdgcn_wmma_f32_16x16x32_bf16(
                false, pa, false, vf, (short)0, o[ns], false, false);
        }
    }

    // ---- epilogue: normalize, write fp32 [B,L,D] ----
#pragma unroll
    for (int ns = 0; ns < 4; ++ns) {
        const int d = ns * 16 + lcol;
#pragma unroll
        for (int r = 0; r < 8; ++r) {
            const int i = q0 + r + 8 * half;
            out[((size_t)b * LSEQ + i) * DMODEL + h * DH + d] = o[ns][r] / l_r[r];
        }
    }
}

// ---------------------------------------------------------------------------
extern "C" void kernel_launch(void* const* d_in, const int* in_sizes, int n_in,
                              void* d_out, int out_size, void* d_ws, size_t ws_size,
                              hipStream_t stream)
{
    const float* X  = (const float*)d_in[0];
    const float* Wq = (const float*)d_in[1];
    const float* bq = (const float*)d_in[2];
    const float* Wk = (const float*)d_in[3];
    const float* bk = (const float*)d_in[4];
    const float* Wv = (const float*)d_in[5];
    const float* bv = (const float*)d_in[6];
    float* out = (float*)d_out;

    // Workspace layout (bf16):
    //   Xb  : [B*L][D]            = 4096*1024          (8 MB)
    //   Wt  : 3 x [D][D]          = 3*1024*1024        (6 MB)
    //   Qs  : [B][H][L][dh]       = 4 M                (8 MB)
    //   Ks  : [B][H][L][dh]       = 4 M                (8 MB)
    //   Vt  : [B][H][dh][L]       = 4 M                (8 MB)
    bf16_t* Xb = (bf16_t*)d_ws;
    bf16_t* Wt = Xb + (size_t)BATCH * LSEQ * DMODEL;
    bf16_t* Qs = Wt + (size_t)3 * DMODEL * DMODEL;
    bf16_t* Ks = Qs + (size_t)BATCH * NHEAD * LSEQ * DH;
    bf16_t* Vt = Ks + (size_t)BATCH * NHEAD * LSEQ * DH;

    const int n4 = (BATCH * LSEQ * DMODEL) / 4;
    cvt_x_kernel<<<1024, 256, 0, stream>>>(X, Xb, n4);

    dim3 gWt(DMODEL / 32, DMODEL / 32, 3);
    wt_kernel<<<gWt, 256, 0, stream>>>(Wq, Wk, Wv, Wt);

    dim3 gProj(DMODEL / 64, (BATCH * LSEQ) / 64, 3);
    qkv_proj_kernel<<<gProj, dim3(128), 0, stream>>>(Xb, Wt, bq, bk, bv,
                                                     Qs, Ks, Vt);

    dim3 gAttn(LSEQ / 64, NHEAD, BATCH);
    local_attn_kernel<<<gAttn, dim3(128), 0, stream>>>(Qs, Ks, Vt, out);
}
